// LimbOrientationPredictor_22316650070935
// MI455X (gfx1250) — compile-verified, matrix operand order checked by reading, not run on hardware
//
#include <hip/hip_runtime.h>
#include <hip/hip_bf16.h>

typedef __attribute__((ext_vector_type(16))) _Float16 v16h;
typedef __attribute__((ext_vector_type(8)))  _Float16 v8h;
typedef __attribute__((ext_vector_type(4)))  _Float16 h4;
typedef __attribute__((ext_vector_type(8)))  float    v8f;
typedef __attribute__((ext_vector_type(4)))  float    f4;

#define BATCH 65536
#define NL 14
#define ROWS (BATCH*NL)        /* 917504 */
#define TILES (ROWS/16)        /* 57344  */
#define WAVES_PER_BLOCK 4
#define THREADS (WAVES_PER_BLOCK*32)
#define NBLOCKS 1792

__device__ __constant__ int c_parent[NL] = {5,7,6,8,11,13,12,14,5,11,5,6,5,6};
__device__ __constant__ int c_child [NL] = {7,9,8,10,13,15,14,16,6,12,11,12,12,11};

__device__ __forceinline__ v16h mk16(v8h lo, v8h hi) {
  v16h r;
#pragma unroll
  for (int i = 0; i < 8; ++i) { r[i] = lo[i]; r[i + 8] = hi[i]; }
  return r;
}

// Fragment gather matching the CDNA5 16-bit 16x32 A layout:
// lane = 16*hh + m holds K runs [k*32 + 8*hh, +8) and [k*32 + 16 + 8*hh, +8).
// Same pattern serves B fragments when `row` points at a weight row (output col).
__device__ __forceinline__ v16h frag16(const _Float16* row, int k, int hh) {
  const v8h a = *(const v8h*)(row + k * 32 + hh * 8);
  const v8h b = *(const v8h*)(row + k * 32 + 16 + hh * 8);
  return mk16(a, b);
}

__device__ __forceinline__ v8f wmma16(v16h a, v16h b, v8f c) {
  return __builtin_amdgcn_wmma_f32_16x16x32_f16(false, a, false, b, (short)0, c, false, false);
}

__device__ __forceinline__ void cfence() { asm volatile("" ::: "memory"); }

extern "C" __global__ void prep_weights(const float* __restrict__ h1_w,
                                        const float* __restrict__ h2_w,
                                        const float* __restrict__ h3_w,
                                        _Float16* __restrict__ w1h,
                                        _Float16* __restrict__ w2h,
                                        _Float16* __restrict__ w3p) {
  int i = blockIdx.x * blockDim.x + threadIdx.x;
  if (i < 32768) w1h[i] = (_Float16)h1_w[i];
  int j = i - 32768;
  if (j >= 0 && j < 16384) w2h[j] = (_Float16)h2_w[j];
  int k = i - 49152;
  if (k >= 0 && k < 2048) {
    int o = k >> 7, kk = k & 127;
    w3p[k] = (o < 3) ? (_Float16)h3_w[o * 128 + kk] : (_Float16)0.0f;
  }
}

extern "C" __global__ void prep_view(const float* __restrict__ az,
                                     const float* __restrict__ el,
                                     const float* __restrict__ v_w,
                                     const float* __restrict__ v_b,
                                     _Float16* __restrict__ vfh) {
  int b = blockIdx.x * blockDim.x + threadIdx.x;
  if (b >= BATCH) return;
  const float d2r = 0.017453292519943295f;
  float a = az[b] * d2r, e = el[b] * d2r;
  float sa = __sinf(a), ca = __cosf(a), se = __sinf(e), ce = __cosf(e);
  float vi[7] = {sa * ce, se, ca * ce, sa, ca, se, ce};
#pragma unroll
  for (int o = 0; o < 32; ++o) {
    float s = v_b[o];
#pragma unroll
    for (int kk = 0; kk < 7; ++kk) s += vi[kk] * v_w[o * 7 + kk];
    vfh[(size_t)b * 32 + o] = (_Float16)fmaxf(s, 0.0f);
  }
}

extern "C" __global__ __launch_bounds__(THREADS, 4)
void limb_main(const float* __restrict__ jf,  const float* __restrict__ p3,
               const float* __restrict__ pr2,
               const float* __restrict__ co_w, const float* __restrict__ co_b,
               const float* __restrict__ co_g, const float* __restrict__ co_be,
               const float* __restrict__ l2_w, const float* __restrict__ l2_b,
               const float* __restrict__ l2_g, const float* __restrict__ l2_be,
               const float* __restrict__ h1_b, const float* __restrict__ h_g,
               const float* __restrict__ h_be, const float* __restrict__ h2_b,
               const float* __restrict__ h3_b,
               const _Float16* __restrict__ w1h, const _Float16* __restrict__ w2h,
               const _Float16* __restrict__ w3p, const _Float16* __restrict__ vfh,
               float* __restrict__ out) {
  // One 16x256 f16 tile per wave. It serves, in sequence:
  //   lf activations -> (in place) hidden activations (cols 0..127) -> delta scratch.
  __shared__ __align__(16) _Float16 s_lf[WAVES_PER_BLOCK][16 * 256];

  const int lane = threadIdx.x & 31;
  const int wid  = threadIdx.x >> 5;
  const int rr = lane & 15;   // row (phase 1 / A-frag) or output column-in-tile (B/D)
  const int hh = lane >> 4;   // wave half
  _Float16* lf = s_lf[wid];

  // Tile-invariant per-lane column constants for h1/h2/h3.
  float hg8[8], hbe8[8], h1b8[8], h2b8[8];
#pragma unroll
  for (int nt = 0; nt < 8; ++nt) {
    int col = nt * 16 + rr;
    hg8[nt]  = h_g[col];
    hbe8[nt] = h_be[col];
    h1b8[nt] = h1_b[col];
    h2b8[nt] = h2_b[col];
  }
  const float h3bv = (rr < 3) ? h3_b[rr] : 0.0f;

  const int gwave  = blockIdx.x * WAVES_PER_BLOCK + wid;
  const int nwaves = gridDim.x * WAVES_PER_BLOCK;

  for (int t = gwave; t < TILES; t += nwaves) {
    const int r = t * 16 + rr;
    const unsigned b = (unsigned)r / NL;
    const int l  = r - (int)b * NL;
    const int jp = c_parent[l], jc = c_child[l];

    // ---- phase 1: per-row features into the 16x256 lf tile -------------
    const float* pp = p3 + ((size_t)b * 17 + jp) * 3;
    const float* pc = p3 + ((size_t)b * 17 + jc) * 3;
    float dx = pc[0] - pp[0], dy = pc[1] - pp[1], dz = pc[2] - pp[2];
    float inv = 1.0f / fmaxf(sqrtf(dx * dx + dy * dy + dz * dz), 1e-6f);
    float cx = dx * inv, cy = dy * inv, cz = dz * inv;

    // co: 32 outputs, each half computes 16; raw values staged in LDS (f16),
    // moments via shuffles, then normalize in place (keeps VGPR arrays dead).
    {
      float s1 = 0.f, s2 = 0.f;
#pragma unroll
      for (int i = 0; i < 16; ++i) {
        int o = hh * 16 + i;
        float a = cx * co_w[o * 3] + cy * co_w[o * 3 + 1] + cz * co_w[o * 3 + 2] + co_b[o];
        lf[rr * 256 + 64 + o] = (_Float16)a;
        s1 += a; s2 += a * a;
      }
      s1 += __shfl_xor(s1, 16); s2 += __shfl_xor(s2, 16);
      float mu = s1 * (1.0f / 32.0f);
      float rs = rsqrtf(s2 * (1.0f / 32.0f) - mu * mu + 1e-5f);
      cfence();
#pragma unroll
      for (int i = 0; i < 16; ++i) {
        int o = hh * 16 + i;
        float x = ((float)lf[rr * 256 + 64 + o] - mu) * rs * co_g[o] + co_be[o];
        lf[rr * 256 + 64 + o] = (_Float16)fmaxf(x, 0.0f);
      }
    }

    // l2: 64 outputs from 4 2D-limb features, LN over 64 (same LDS staging)
    {
      const float* qp = pr2 + ((size_t)b * 17 + jp) * 2;
      const float* qc = pr2 + ((size_t)b * 17 + jc) * 2;
      float vx = qc[0] - qp[0], vy = qc[1] - qp[1];
      float L = sqrtf(vx * vx + vy * vy);
      float il = 1.0f / (L + 1e-6f);
      float f0 = vx * il, f1 = vy * il, f2 = L, f3 = L / (0.15f + 1e-6f);
      float s1 = 0.f, s2 = 0.f;
#pragma unroll
      for (int i = 0; i < 32; ++i) {
        int o = hh * 32 + i;
        float a = f0 * l2_w[o * 4] + f1 * l2_w[o * 4 + 1] + f2 * l2_w[o * 4 + 2] +
                  f3 * l2_w[o * 4 + 3] + l2_b[o];
        lf[rr * 256 + o] = (_Float16)a;
        s1 += a; s2 += a * a;
      }
      s1 += __shfl_xor(s1, 16); s2 += __shfl_xor(s2, 16);
      float mu = s1 * (1.0f / 64.0f);
      float rs = rsqrtf(s2 * (1.0f / 64.0f) - mu * mu + 1e-5f);
      cfence();
#pragma unroll
      for (int i = 0; i < 32; ++i) {
        int o = hh * 32 + i;
        float x = ((float)lf[rr * 256 + o] - mu) * rs * l2_g[o] + l2_be[o];
        lf[rr * 256 + o] = (_Float16)fmaxf(x, 0.0f);
      }
    }

    // pf / cf: 64 f32 each -> f16, halves split the columns
    {
      const f4* sp = (const f4*)(jf + ((size_t)b * 17 + jp) * 64 + hh * 32);
      const f4* sc = (const f4*)(jf + ((size_t)b * 17 + jc) * 64 + hh * 32);
      _Float16* dp = lf + rr * 256 + 96  + hh * 32;
      _Float16* dc = lf + rr * 256 + 160 + hh * 32;
#pragma unroll
      for (int q = 0; q < 8; ++q) {
        f4 a = sp[q], c2 = sc[q];
        h4 oa, oc;
#pragma unroll
        for (int e = 0; e < 4; ++e) { oa[e] = (_Float16)a[e]; oc[e] = (_Float16)c2[e]; }
        *(h4*)(dp + 4 * q) = oa;
        *(h4*)(dc + 4 * q) = oc;
      }
    }

    // vf: precomputed per-batch view features (f16 copy)
    {
      const v8h* sv = (const v8h*)(vfh + (size_t)b * 32 + hh * 16);
      *(v8h*)(lf + rr * 256 + 224 + hh * 16)     = sv[0];
      *(v8h*)(lf + rr * 256 + 224 + hh * 16 + 8) = sv[1];
    }
    cfence(); // LDS is in-order per wave; just block compiler reordering

    // ---- h1: 16x256 @ 256x128, k-outer, accumulators start from inline-0 C
    v8f acc[8];
#pragma unroll
    for (int nt = 0; nt < 8; ++nt) acc[nt] = (v8f){};
#pragma unroll
    for (int k = 0; k < 8; ++k) {
      const v16h a = frag16(lf + rr * 256, k, hh);
#pragma unroll
      for (int nt = 0; nt < 8; ++nt) {
        const _Float16* wrow = w1h + (size_t)(nt * 16 + rr) * 256;
        acc[nt] = wmma16(a, frag16(wrow, k, hh), acc[nt]);
      }
    }

    // Bias folded here; row-wise LN over 128 (row m = 8*hh + j)
    float mu[8], rs[8];
#pragma unroll
    for (int j = 0; j < 8; ++j) {
      float s1 = 0.f, s2 = 0.f;
#pragma unroll
      for (int nt = 0; nt < 8; ++nt) {
        float x = acc[nt][j] + h1b8[nt];
        s1 += x; s2 += x * x;
      }
#pragma unroll
      for (int m = 1; m < 16; m <<= 1) { s1 += __shfl_xor(s1, m); s2 += __shfl_xor(s2, m); }
      float m1 = s1 * (1.0f / 128.0f);
      mu[j] = m1;
      rs[j] = rsqrtf(s2 * (1.0f / 128.0f) - m1 * m1 + 1e-5f);
    }
    // hidden activations overwrite lf cols [0,128) (lf fully consumed by h1)
#pragma unroll
    for (int nt = 0; nt < 8; ++nt)
#pragma unroll
      for (int j = 0; j < 8; ++j) {
        float x = (acc[nt][j] + h1b8[nt] - mu[j]) * rs[j] * hg8[nt] + hbe8[nt];
        x = x > 0.f ? x : 0.2f * x;
        lf[(8 * hh + j) * 256 + nt * 16 + rr] = (_Float16)x;
      }
    cfence();

    // ---- h2: 16x128 @ 128x128, k-outer, inline-0 C ---------------------
#pragma unroll
    for (int nt = 0; nt < 8; ++nt) acc[nt] = (v8f){};
#pragma unroll
    for (int k = 0; k < 4; ++k) {
      const v16h a = frag16(lf + rr * 256, k, hh);
#pragma unroll
      for (int nt = 0; nt < 8; ++nt) {
        const _Float16* wrow = w2h + (size_t)(nt * 16 + rr) * 128;
        acc[nt] = wmma16(a, frag16(wrow, k, hh), acc[nt]);
      }
    }
#pragma unroll
    for (int nt = 0; nt < 8; ++nt)
#pragma unroll
      for (int j = 0; j < 8; ++j) {
        float x = acc[nt][j] + h2b8[nt];
        x = x > 0.f ? x : 0.2f * x;
        lf[(8 * hh + j) * 256 + nt * 16 + rr] = (_Float16)x;
      }
    cfence();

    // ---- h3: 16x128 @ 128x16(pad, 3 valid) via 4 WMMA, inline-0 C ------
    v8f c3 = (v8f){};
    {
      const _Float16* wrow = w3p + (size_t)rr * 128;
#pragma unroll
      for (int k = 0; k < 4; ++k)
        c3 = wmma16(frag16(lf + rr * 256, k, hh), frag16(wrow, k, hh), c3);
    }

    // ---- finalize: delta -> LDS, lanes 0..15 normalize their row -------
    float* db = (float*)lf; // tile fully consumed; reuse as 16x4 f32 delta buffer
    cfence();
    if (rr < 3) {
#pragma unroll
      for (int j = 0; j < 8; ++j) db[(8 * hh + j) * 4 + rr] = c3[j] + h3bv;
    }
    cfence();
    if (hh == 0) {
      float ox = cx + db[rr * 4 + 0];
      float oy = cy + db[rr * 4 + 1];
      float oz = cz + db[rr * 4 + 2];
      float iv = 1.0f / fmaxf(sqrtf(ox * ox + oy * oy + oz * oz), 1e-6f);
      float* po = out + (size_t)(t * 16 + rr) * 3;
      po[0] = ox * iv; po[1] = oy * iv; po[2] = oz * iv;
    }
    cfence();
  }
}

extern "C" void kernel_launch(void* const* d_in, const int* in_sizes, int n_in,
                              void* d_out, int out_size, void* d_ws, size_t ws_size,
                              hipStream_t stream) {
  const float* jf    = (const float*)d_in[0];
  /* d_in[1] pose_2d is unused by the reference */
  const float* p3    = (const float*)d_in[2];
  const float* az    = (const float*)d_in[3];
  const float* el    = (const float*)d_in[4];
  const float* pr2   = (const float*)d_in[5];
  const float* co_w  = (const float*)d_in[6];
  const float* co_b  = (const float*)d_in[7];
  const float* co_g  = (const float*)d_in[8];
  const float* co_be = (const float*)d_in[9];
  const float* l2_w  = (const float*)d_in[10];
  const float* l2_b  = (const float*)d_in[11];
  const float* l2_g  = (const float*)d_in[12];
  const float* l2_be = (const float*)d_in[13];
  const float* v_w   = (const float*)d_in[14];
  const float* v_b   = (const float*)d_in[15];
  const float* h1_w  = (const float*)d_in[16];
  const float* h1_b  = (const float*)d_in[17];
  const float* h_g   = (const float*)d_in[18];
  const float* h_be  = (const float*)d_in[19];
  const float* h2_w  = (const float*)d_in[20];
  const float* h2_b  = (const float*)d_in[21];
  const float* h3_w  = (const float*)d_in[22];
  const float* h3_b  = (const float*)d_in[23];

  _Float16* w1h = (_Float16*)d_ws;      // 32768
  _Float16* w2h = w1h + 32768;          // 16384
  _Float16* w3p = w2h + 16384;          // 2048 (16x128, rows>=3 zero)
  _Float16* vfh = w3p + 2048;           // BATCH*32

  prep_weights<<<200, 256, 0, stream>>>(h1_w, h2_w, h3_w, w1h, w2h, w3p);
  prep_view<<<BATCH / 256, 256, 0, stream>>>(az, el, v_w, v_b, vfh);
  limb_main<<<NBLOCKS, THREADS, 0, stream>>>(jf, p3, pr2, co_w, co_b, co_g, co_be,
                                             l2_w, l2_b, l2_g, l2_be,
                                             h1_b, h_g, h_be, h2_b, h3_b,
                                             w1h, w2h, w3p, vfh, (float*)d_out);
}
